// HeteroGATLayer_85461259256043
// MI455X (gfx1250) — matrix operand order repaired and checked by reading, not compiled, not run on hardware
//
#include <hip/hip_runtime.h>

// HeteroGAT layer for MI455X (gfx1250, wave32, WMMA).
// N = 50000, E = 800000, DIM = 128, H = 4 heads, HD = 32, NET = 7.
// GEMMs (6.6 GFLOP) -> v_wmma_f32_16x16x32_f16 (f16 in / f32 accumulate).
// Edge gather/scatter is L2-resident (Q/K/V + agg ~103 MB < 192 MB L2), so the
// softmax-aggregate runs as coalesced f32 loads + global_atomic_add_f32 in L2.

typedef __attribute__((ext_vector_type(16))) _Float16 v16h;
typedef __attribute__((ext_vector_type(8)))  _Float16 v8h;
typedef __attribute__((ext_vector_type(8)))  float    v8f;

#define DIM 128
#define NH 4
#define HDIM 32
#define NTYPES 7

// ---------- helpers ----------

__device__ __forceinline__ unsigned fenc(float f) {
    unsigned u = __float_as_uint(f);
    return u ^ (unsigned)(((int)u >> 31) | 0x80000000);
}
__device__ __forceinline__ float fdec(unsigned u) {
    unsigned b = u ^ (((int)u >= 0) ? 0xFFFFFFFFu : 0x80000000u);
    return __uint_as_float(b);
}

// A-fragment (16x32 f16): lane%16 = row M; lanes 0-15 hold K = kk*32+[0..16),
// lanes 16-31 hold K = kk*32+[16..32). Two 16B loads per lane.
__device__ __forceinline__ v16h load_a_frag(const _Float16* __restrict__ Ah,
                                            int node0, int lane, int kk, int N) {
    int r = node0 + (lane & 15);
    r = (r < N) ? r : (N - 1);                 // clamp (no branch; EXEC stays full)
    const int kb = kk * 32 + ((lane >> 4) << 4);
    const _Float16* p = Ah + (size_t)r * DIM + kb;
    v8h lo = *(const v8h*)(p);
    v8h hi = *(const v8h*)(p + 8);
    v16h a;
#pragma unroll
    for (int i = 0; i < 8; ++i) { a[i] = lo[i]; a[8 + i] = hi[i]; }
    return a;
}

// B-fragment (32x16 f16) from TRANSPOSED weights Wt[n][k]: lane%16 = col N;
// lanes 0-15 hold K = kk*32+[0..16), lanes 16-31 hold K = kk*32+[16..32).
// One contiguous 32B load per lane.
__device__ __forceinline__ v16h load_b_frag(const _Float16* __restrict__ Wt,
                                            int colBase, int lane, int kk) {
    const int n = colBase + (lane & 15);
    const int kb = kk * 32 + ((lane >> 4) << 4);
    return *(const v16h*)(Wt + n * DIM + kb);
}

// ---------- prep kernels ----------

__global__ void k_f2h(const float* __restrict__ src, _Float16* __restrict__ dst, int n) {
    int i = blockIdx.x * blockDim.x + threadIdx.x;
    if (i < n) dst[i] = (_Float16)src[i];
}

__global__ void k_prep_weights(const float* __restrict__ Wq, const float* __restrict__ Wk,
                               const float* __restrict__ Wv, const float* __restrict__ Wo,
                               _Float16* __restrict__ Wqt, _Float16* __restrict__ Wkt,
                               _Float16* __restrict__ Wvt, _Float16* __restrict__ Wot) {
    int idx = blockIdx.x * blockDim.x + threadIdx.x;
    if (idx >= 4 * DIM * DIM) return;
    int m = idx >> 14;
    int rem = idx & (DIM * DIM - 1);
    int n = rem >> 7, k = rem & 127;
    const float* s = (m == 0) ? Wq : (m == 1) ? Wk : (m == 2) ? Wv : Wo;
    _Float16* d = (m == 0) ? Wqt : (m == 1) ? Wkt : (m == 2) ? Wvt : Wot;
    d[n * DIM + k] = (_Float16)s[k * DIM + n];   // Wt[n][k] = W[k][n]
}

__global__ void k_zero_f32(float* __restrict__ p, int n) {
    int i = blockIdx.x * blockDim.x + threadIdx.x;
    if (i < n) p[i] = 0.0f;
}

__global__ void k_init_gmax(unsigned* __restrict__ g) {
    if (blockIdx.x == 0 && threadIdx.x == 0) *g = 0u;  // encoded minimum
}

// ---------- QKV GEMM (WMMA) ----------
// One workgroup = 8 waves per 16-node tile; A frags reused across 3 matrices.
// Tile-bound check is hoisted to one uniform scalar branch so the hot path has
// no exec-mask juggling around the stores (EXEC stays all-ones for WMMA).

__global__ void k_qkv(const _Float16* __restrict__ Xh,
                      const _Float16* __restrict__ Wqt,
                      const _Float16* __restrict__ Wkt,
                      const _Float16* __restrict__ Wvt,
                      float* __restrict__ Q, float* __restrict__ Kd,
                      float* __restrict__ V, int N) {
    const int node0 = blockIdx.x * 16;
    const int lane = threadIdx.x & 31;
    const int wv = threadIdx.x >> 5;
    const int colBase = wv << 4;
    const bool full = (node0 + 16 <= N);       // uniform across the workgroup

    v16h afr[4];
#pragma unroll
    for (int kk = 0; kk < 4; ++kk) afr[kk] = load_a_frag(Xh, node0, lane, kk, N);

    const _Float16* wts[3] = {Wqt, Wkt, Wvt};
    float* outs[3] = {Q, Kd, V};
    const int col = colBase + (lane & 15);
    const int rbase = node0 + ((lane >> 4) << 3);

#pragma unroll
    for (int mat = 0; mat < 3; ++mat) {
        const _Float16* W = wts[mat];
        v8f c = {};
#pragma unroll
        for (int kk = 0; kk < 4; ++kk) {
            v16h b = load_b_frag(W, colBase, lane, kk);
            c = __builtin_amdgcn_wmma_f32_16x16x32_f16(false, afr[kk], false, b,
                                                       (short)0, c, false, false);
        }
        float* O = outs[mat] + (size_t)rbase * DIM + col;
        if (full) {
#pragma unroll
            for (int j = 0; j < 8; ++j) O[(size_t)j * DIM] = c[j];
        } else {
#pragma unroll
            for (int j = 0; j < 8; ++j)
                if (rbase + j < N) O[(size_t)j * DIM] = c[j];
        }
    }
}

// ---------- per-node / per-type score partials ----------

__global__ void k_node_scores(const float* __restrict__ Q, const float* __restrict__ K,
                              const float* __restrict__ av,
                              float* __restrict__ sq, float* __restrict__ sk, int N) {
    int idx = blockIdx.x * blockDim.x + threadIdx.x;
    if (idx >= N * NH) return;
    int n = idx >> 2, h = idx & 3;
    const float* qp = Q + (size_t)n * DIM + h * HDIM;
    const float* kp = K + (size_t)n * DIM + h * HDIM;
    const float* aq = av + h * (3 * HDIM);
    const float* ak = aq + HDIM;
    float s1 = 0.f, s2 = 0.f;
#pragma unroll
    for (int i = 0; i < HDIM; ++i) { s1 += qp[i] * aq[i]; s2 += kp[i] * ak[i]; }
    sq[idx] = s1; sk[idx] = s2;
}

__global__ void k_type_scores(const float* __restrict__ emb, const float* __restrict__ av,
                              float* __restrict__ se) {
    int idx = threadIdx.x;
    if (idx >= NTYPES * NH) return;
    int t = idx >> 2, h = idx & 3;
    const float* ep = emb + t * DIM + h * HDIM;
    const float* ae = av + h * (3 * HDIM) + 2 * HDIM;
    float s = 0.f;
#pragma unroll
    for (int i = 0; i < HDIM; ++i) s += ep[i] * ae[i];
    se[idx] = s;
}

// ---------- edge pass 1: raw scores + global max ----------

__global__ void k_edge_score(const long long* __restrict__ srcI,
                             const long long* __restrict__ dstI,
                             const long long* __restrict__ etype,
                             const float* __restrict__ ew,
                             const float* __restrict__ sq, const float* __restrict__ sk,
                             const float* __restrict__ se,
                             float* __restrict__ scores, unsigned* __restrict__ gmax, int E) {
    __shared__ float smax[256];
    int e = blockIdx.x * blockDim.x + threadIdx.x;
    float lm = -3.4e38f;
    if (e < E) {
        int s = (int)srcI[e], d = (int)dstI[e], t = (int)etype[e];
        float lw = log1pf(ew[e]);
#pragma unroll
        for (int h = 0; h < NH; ++h) {
            float x = sq[d * NH + h] + sk[s * NH + h] + se[t * NH + h];
            x = (x > 0.f) ? x : 0.2f * x;     // leaky_relu(0.2)
            x += lw;
            scores[(size_t)e * NH + h] = x;
            lm = fmaxf(lm, x);
        }
    }
    smax[threadIdx.x] = lm;
    __syncthreads();
    for (int off = 128; off > 0; off >>= 1) {
        if (threadIdx.x < (unsigned)off)
            smax[threadIdx.x] = fmaxf(smax[threadIdx.x], smax[threadIdx.x + off]);
        __syncthreads();
    }
    if (threadIdx.x == 0) atomicMax(gmax, fenc(smax[0]));
}

// ---------- edge pass 2: exp + segment sum ----------

__global__ void k_edge_exp(const long long* __restrict__ dstI,
                           float* __restrict__ scores, float* __restrict__ attn_sum,
                           const unsigned* __restrict__ gmax, int E) {
    int e = blockIdx.x * blockDim.x + threadIdx.x;
    if (e >= E) return;
    const float M = fdec(*gmax);
    const int d = (int)dstI[e];
#pragma unroll
    for (int h = 0; h < NH; ++h) {
        float x = expf(scores[(size_t)e * NH + h] - M);
        scores[(size_t)e * NH + h] = x;
        atomicAdd(&attn_sum[d * NH + h], x);
    }
}

// ---------- edge pass 3: weighted aggregation (one wave per edge) ----------

__global__ void k_edge_agg(const long long* __restrict__ srcI,
                           const long long* __restrict__ dstI,
                           const float* __restrict__ scores,
                           const float* __restrict__ attn_sum,
                           const float* __restrict__ V, float* __restrict__ agg, int E) {
    const int gw = (int)((blockIdx.x * (unsigned)blockDim.x + threadIdx.x) >> 5);
    const int lane = threadIdx.x & 31;
    if (gw >= E) return;
    const int s = (int)srcI[gw], d = (int)dstI[gw];
    __builtin_prefetch(V + (size_t)s * DIM + lane, 0, 0);   // global_prefetch_b8
#pragma unroll
    for (int h = 0; h < NH; ++h) {
        const float coef = scores[(size_t)gw * NH + h] / (attn_sum[d * NH + h] + 1e-9f);
        const int dim = h * HDIM + lane;
        atomicAdd(&agg[(size_t)d * DIM + dim], coef * V[(size_t)s * DIM + dim]);
    }
}

// ---------- output GEMM (WMMA) fused with ELU + residual + LayerNorm ----------

__global__ void k_out(const _Float16* __restrict__ Ah, const _Float16* __restrict__ Wot,
                      const float* __restrict__ Xf, const float* __restrict__ b_o,
                      const float* __restrict__ gamma, const float* __restrict__ beta,
                      float* __restrict__ Out, int N) {
    __shared__ float tile[16][DIM];
    __shared__ float redS[16][16];
    __shared__ float redQ[16][16];
    __shared__ float muS[16], rstdS[16];

    const int node0 = blockIdx.x * 16;
    const int lane = threadIdx.x & 31;
    const int wv = threadIdx.x >> 5;
    const int colBase = wv << 4;

    v8f c = {};
#pragma unroll
    for (int kk = 0; kk < 4; ++kk) {
        v16h a = load_a_frag(Ah, node0, lane, kk, N);
        v16h b = load_b_frag(Wot, colBase, lane, kk);
        c = __builtin_amdgcn_wmma_f32_16x16x32_f16(false, a, false, b,
                                                   (short)0, c, false, false);
    }

    const int col = colBase + (lane & 15);
    const float bias = b_o[col];
    const int rbase = (lane >> 4) << 3;
#pragma unroll
    for (int j = 0; j < 8; ++j) {
        const int row = rbase + j;
        int nrow = node0 + row;
        int crow = (nrow < N) ? nrow : (N - 1);        // clamp read, no branch
        float g = c[j] + bias;
        g = (g > 0.f) ? g : (expf(g) - 1.0f);          // ELU
        g += Xf[(size_t)crow * DIM + col];             // residual
        tile[row][col] = g;
    }
    __syncthreads();

    // LayerNorm: 16 threads per row, 8 elements per thread
    const int row = threadIdx.x >> 4;
    const int seg = threadIdx.x & 15;
    float s = 0.f, q = 0.f;
#pragma unroll
    for (int i = 0; i < 8; ++i) {
        float v = tile[row][seg * 8 + i];
        s += v; q += v * v;
    }
    redS[row][seg] = s; redQ[row][seg] = q;
    __syncthreads();
    if (seg == 0) {
        float ts = 0.f, tq = 0.f;
#pragma unroll
        for (int i = 0; i < 16; ++i) { ts += redS[row][i]; tq += redQ[row][i]; }
        const float m = ts * (1.0f / DIM);
        const float var = tq * (1.0f / DIM) - m * m;
        muS[row] = m;
        rstdS[row] = rsqrtf(var + 1e-5f);
    }
    __syncthreads();
    const float m = muS[row], r = rstdS[row];
    const int orow = node0 + row;
    if (orow < N) {                                     // uniform per 16-thread row group
#pragma unroll
        for (int i = 0; i < 8; ++i) {
            const int cc = seg * 8 + i;
            float v = (tile[row][cc] - m) * r;
            Out[(size_t)orow * DIM + cc] = v * gamma[cc] + beta[cc];
        }
    }
}

// ---------- launch ----------

extern "C" void kernel_launch(void* const* d_in, const int* in_sizes, int n_in,
                              void* d_out, int out_size, void* d_ws, size_t ws_size,
                              hipStream_t stream) {
    const float*     X   = (const float*)d_in[0];
    const long long* ei  = (const long long*)d_in[1];
    const long long* et  = (const long long*)d_in[2];
    const float*     ew  = (const float*)d_in[3];
    const float*     Wq  = (const float*)d_in[4];
    const float*     Wk  = (const float*)d_in[5];
    const float*     Wv  = (const float*)d_in[6];
    const float*     emb = (const float*)d_in[7];
    const float*     av  = (const float*)d_in[8];
    const float*     Wo  = (const float*)d_in[9];
    const float*     bo  = (const float*)d_in[10];
    const float*     gam = (const float*)d_in[11];
    const float*     bet = (const float*)d_in[12];
    float* Out = (float*)d_out;

    const int N = in_sizes[0] / DIM;
    const int E = in_sizes[3];
    const long long* srcI = ei;
    const long long* dstI = ei + E;

    // workspace layout (256B aligned), total ~143 MB
    char* ws = (char*)d_ws;
    size_t off = 0;
    auto alloc = [&](size_t bytes) -> void* {
        void* p = ws + off;
        off = (off + bytes + 255) & ~(size_t)255;
        return p;
    };
    _Float16* Xh   = (_Float16*)alloc((size_t)N * DIM * sizeof(_Float16));
    _Float16* Wqt  = (_Float16*)alloc(DIM * DIM * sizeof(_Float16));
    _Float16* Wkt  = (_Float16*)alloc(DIM * DIM * sizeof(_Float16));
    _Float16* Wvt  = (_Float16*)alloc(DIM * DIM * sizeof(_Float16));
    _Float16* Wot  = (_Float16*)alloc(DIM * DIM * sizeof(_Float16));
    float*    Q    = (float*)alloc((size_t)N * DIM * sizeof(float));
    float*    Kd   = (float*)alloc((size_t)N * DIM * sizeof(float));
    float*    V    = (float*)alloc((size_t)N * DIM * sizeof(float));
    float*    sq   = (float*)alloc((size_t)N * NH * sizeof(float));
    float*    sk   = (float*)alloc((size_t)N * NH * sizeof(float));
    float*    se   = (float*)alloc(NTYPES * NH * sizeof(float));
    float*    scores   = (float*)alloc((size_t)E * NH * sizeof(float));
    float*    attn_sum = (float*)alloc((size_t)N * NH * sizeof(float));
    float*    agg  = (float*)alloc((size_t)N * DIM * sizeof(float));
    _Float16* aggh = (_Float16*)alloc((size_t)N * DIM * sizeof(_Float16));
    unsigned* gmax = (unsigned*)alloc(sizeof(unsigned));
    (void)ws_size; (void)n_in; (void)out_size;

    const int B = 256;
    const int ntiles = (N + 15) / 16;

    // prep
    k_f2h<<<(N * DIM + B - 1) / B, B, 0, stream>>>(X, Xh, N * DIM);
    k_prep_weights<<<(4 * DIM * DIM + B - 1) / B, B, 0, stream>>>(
        Wq, Wk, Wv, Wo, Wqt, Wkt, Wvt, Wot);
    k_zero_f32<<<(N * DIM + B - 1) / B, B, 0, stream>>>(agg, N * DIM);
    k_zero_f32<<<(N * NH + B - 1) / B, B, 0, stream>>>(attn_sum, N * NH);
    k_init_gmax<<<1, 32, 0, stream>>>(gmax);

    // QKV GEMMs (WMMA)
    k_qkv<<<ntiles, B, 0, stream>>>(Xh, Wqt, Wkt, Wvt, Q, Kd, V, N);

    // score partials
    k_node_scores<<<(N * NH + B - 1) / B, B, 0, stream>>>(Q, Kd, av, sq, sk, N);
    k_type_scores<<<1, 32, 0, stream>>>(emb, av, se);

    // edge passes
    k_edge_score<<<(E + B - 1) / B, B, 0, stream>>>(srcI, dstI, et, ew, sq, sk, se,
                                                    scores, gmax, E);
    k_edge_exp<<<(E + B - 1) / B, B, 0, stream>>>(dstI, scores, attn_sum, gmax, E);
    k_edge_agg<<<(E + 7) / 8, B, 0, stream>>>(srcI, dstI, scores, attn_sum, V, agg, E);

    // output GEMM + epilogue
    k_f2h<<<(N * DIM + B - 1) / B, B, 0, stream>>>(agg, aggh, N * DIM);
    k_out<<<ntiles, B, 0, stream>>>(aggh, Wot, X, bo, gam, bet, Out, N);
}
